// MultiReferFuser_84284438217103
// MI455X (gfx1250) — matrix operand order repaired and checked by reading, not compiled, not run on hardware
//
#include <hip/hip_runtime.h>
#include <hip/hip_bf16.h>

#define NREF 4096
#define EDIM 384
#define HEADS 6
#define DH 64
#define MIDD 96
#define QKVD (3 * EDIM)   // 1152

typedef __bf16 v16bf __attribute__((ext_vector_type(16)));
typedef float  v8f   __attribute__((ext_vector_type(8)));

// ---------------------------------------------------------------------------
// WMMA helpers (CDNA5 wave32, 16x16x32 bf16 -> f32)
// ---------------------------------------------------------------------------
__device__ __forceinline__ v8f wmma_bf16(v16bf a, v16bf b, v8f c) {
  return __builtin_amdgcn_wmma_f32_16x16x32_bf16(
      /*neg_a=*/false, a, /*neg_b=*/false, b,
      /*c_mod=*/(short)0, c, /*reuse_a=*/false, /*reuse_b=*/false);
}

union FragU { v16bf v; uint4 u[2]; };

// A fragment: 16x32 bf16 tile from row-major matrix p (ld elems per row).
__device__ __forceinline__ v16bf load_a_frag(const __bf16* __restrict__ p,
                                             int ld, int r0, int kb) {
  int lane = threadIdx.x & 31;
  int row = r0 + (lane & 15);
  int kg = lane >> 4;
  const __bf16* base = p + (size_t)row * ld + kb + kg * 8;
  FragU f;
  f.u[0] = *(const uint4*)(base);
  f.u[1] = *(const uint4*)(base + 16);
  return f.v;
}

// B fragment from B^T stored row-major: element (k, col) at p[col*ld + k].
__device__ __forceinline__ v16bf load_bT_frag(const __bf16* __restrict__ p,
                                              int ld, int c0, int kb) {
  int lane = threadIdx.x & 31;
  int col = c0 + (lane & 15);
  int kg = lane >> 4;
  const __bf16* base = p + (size_t)col * ld + kb + kg * 8;
  FragU f;
  f.u[0] = *(const uint4*)(base);
  f.u[1] = *(const uint4*)(base + 16);
  return f.v;
}

// Async global -> LDS copy, 16 bytes per lane (ASYNCcnt tracked).
__device__ __forceinline__ void async_load_b128(unsigned lds_off, const void* gaddr) {
  asm volatile("global_load_async_to_lds_b128 %0, %1, off"
               :: "v"(lds_off), "v"((unsigned long long)(size_t)gaddr)
               : "memory");
}
__device__ __forceinline__ void wait_async0() {
  asm volatile("s_wait_asynccnt 0x0" ::: "memory");
}

// ---------------------------------------------------------------------------
// Kernel 0a: f32 -> bf16 conversion (same layout)
// ---------------------------------------------------------------------------
__global__ void cvt_bf16_kernel(const float* __restrict__ in,
                                __bf16* __restrict__ out, int n) {
  int i = blockIdx.x * blockDim.x + threadIdx.x;
  if (i < n) out[i] = (__bf16)in[i];
}

// Kernel 0b: f32 [K x N] row-major -> bf16 B^T [N x K] row-major
__global__ void cvt_bf16_t_kernel(const float* __restrict__ in,
                                  __bf16* __restrict__ out, int K, int N) {
  int i = blockIdx.x * blockDim.x + threadIdx.x;
  if (i < K * N) {
    int k = i / N, n = i % N;
    out[(size_t)n * K + k] = (__bf16)in[i];
  }
}

// ---------------------------------------------------------------------------
// Kernel 1: channel MLPs + softmax fuse. One wave per 16-row tile.
// blockDim = 64 (2 waves), grid = NREF/32
// ---------------------------------------------------------------------------
__global__ void mlp_fuse_kernel(const float* __restrict__ low,
                                const float* __restrict__ high,
                                const __bf16* __restrict__ lowb,
                                const __bf16* __restrict__ highb,
                                const __bf16* __restrict__ w1lt, const float* __restrict__ b1l,
                                const __bf16* __restrict__ w2lt, const float* __restrict__ b2l,
                                const __bf16* __restrict__ w1ht, const float* __restrict__ b1h,
                                const __bf16* __restrict__ w2ht, const float* __restrict__ b2h,
                                const float* __restrict__ layer_fuse,
                                float* __restrict__ fused_f32,
                                __bf16* __restrict__ fused_b16) {
  __shared__ __attribute__((aligned(32))) __bf16 lds_h[2][16 * MIDD];
  int wave = threadIdx.x >> 5;
  int lane = threadIdx.x & 31;
  int tile = blockIdx.x * 2 + wave;
  int r0 = tile * 16;
  int half = lane >> 4, cil = lane & 15;

  float f0 = layer_fuse[0], f1 = layer_fuse[1];
  float mx = fmaxf(f0, f1);
  float e0 = __expf(f0 - mx), e1 = __expf(f1 - mx);
  float lw0 = e0 / (e0 + e1), lw1 = e1 / (e0 + e1);

  __bf16* hbuf = lds_h[wave];
  const v8f vzero = {};

  for (int pass = 0; pass < 2; ++pass) {
    const __bf16* xb = pass ? highb : lowb;
    const float*  xf = pass ? high  : low;
    const __bf16* w1 = pass ? w1ht : w1lt;
    const float*  b1 = pass ? b1h : b1l;
    const __bf16* w2 = pass ? w2ht : w2lt;
    const float*  b2 = pass ? b2h : b2l;
    float lw = pass ? lw1 : lw0;

    // GEMM1: h = x[16x384] @ w1[384x96]   (B from w1^T, ld = EDIM)
    v8f acc[6];
#pragma unroll
    for (int t = 0; t < 6; ++t) acc[t] = vzero;
    for (int kc = 0; kc < EDIM; kc += 32) {
      v16bf a = load_a_frag(xb, EDIM, r0, kc);
#pragma unroll
      for (int t = 0; t < 6; ++t) {
        v16bf b = load_bT_frag(w1, EDIM, t * 16, kc);
        acc[t] = wmma_bf16(a, b, acc[t]);
      }
    }
#pragma unroll
    for (int t = 0; t < 6; ++t) {
      int col = t * 16 + cil;
      float bias = b1[col];
#pragma unroll
      for (int r = 0; r < 8; ++r) {
        float v = acc[t][r] + bias;
        float g = 0.5f * v * (1.0f + erff(v * 0.70710678118f));
        hbuf[(r + 8 * half) * MIDD + col] = (__bf16)g;
      }
    }
    __syncthreads();

    // GEMM2: y = x * sigmoid(h[16x96] @ w2[96x384] + b2)  (B from w2^T, ld = MIDD)
    for (int ct = 0; ct < 24; ++ct) {
      v8f c = vzero;
#pragma unroll
      for (int kc = 0; kc < MIDD; kc += 32) {
        v16bf a = load_a_frag(hbuf, MIDD, 0, kc);
        v16bf b = load_bT_frag(w2, MIDD, ct * 16, kc);
        c = wmma_bf16(a, b, c);
      }
      int col = ct * 16 + cil;
      float bias = b2[col];
#pragma unroll
      for (int r = 0; r < 8; ++r) {
        int row = r0 + r + 8 * half;
        size_t idx = (size_t)row * EDIM + col;
        float s = 1.0f / (1.0f + __expf(-(c[r] + bias)));
        float y = lw * xf[idx] * s;
        if (pass == 0) {
          fused_f32[idx] = y;
        } else {
          float f = fused_f32[idx] + y;
          fused_f32[idx] = f;
          fused_b16[idx] = (__bf16)f;
        }
      }
    }
    __syncthreads();
  }
}

// ---------------------------------------------------------------------------
// Kernel 2: QKV projection. Q,K row-major; V transposed into vT [H*DH x NREF].
// One wave computes a 16x64 tile. blockDim=128, grid = 256*18/4
// ---------------------------------------------------------------------------
__global__ void qkv_kernel(const __bf16* __restrict__ A,
                           const __bf16* __restrict__ Bt,
                           const float* __restrict__ bias,
                           __bf16* __restrict__ out,
                           __bf16* __restrict__ vT) {
  int wave = threadIdx.x >> 5, lane = threadIdx.x & 31;
  int task = blockIdx.x * 4 + wave;
  int rt = task / 18, cc = task % 18;
  int r0 = rt * 16, c0 = cc * 64;
  int half = lane >> 4, cil = lane & 15;
  const v8f vzero = {};
  v8f acc[4];
#pragma unroll
  for (int t = 0; t < 4; ++t) acc[t] = vzero;
  for (int kc = 0; kc < EDIM; kc += 32) {
    v16bf a = load_a_frag(A, EDIM, r0, kc);
#pragma unroll
    for (int t = 0; t < 4; ++t) {
      v16bf b = load_bT_frag(Bt, EDIM, c0 + t * 16, kc);
      acc[t] = wmma_bf16(a, b, acc[t]);
    }
  }
#pragma unroll
  for (int t = 0; t < 4; ++t) {
    int col = c0 + t * 16 + cil;
    float bs = bias[col];
#pragma unroll
    for (int r = 0; r < 8; ++r) {
      int row = r0 + r + 8 * half;
      float v = acc[t][r] + bs;
      if (col < 2 * EDIM) {
        out[(size_t)row * QKVD + col] = (__bf16)v;               // Q,K row-major
      } else {
        vT[(size_t)(col - 2 * EDIM) * NREF + row] = (__bf16)v;   // V transposed
      }
    }
  }
}

// ---------------------------------------------------------------------------
// Kernel 3: flash attention. One wave per (head, 16-query tile); the 4 waves
// of a block share one head, so K/V^T blocks are staged once per block into
// LDS via async global->LDS (ASYNCcnt) and consumed by all 4 waves.
// blockDim=128, grid = HEADS*256/4
// ---------------------------------------------------------------------------
__global__ void attn_kernel(const __bf16* __restrict__ qkv,
                            const __bf16* __restrict__ vT,
                            __bf16* __restrict__ ctx) {
  __shared__ __attribute__((aligned(64))) __bf16 ldsK[32 * DH];   // [key][d]
  __shared__ __attribute__((aligned(64))) __bf16 ldsV[DH * 32];   // [d][key]
  __shared__ __attribute__((aligned(32))) __bf16 ldsP[4][16 * 32];
  int tid = threadIdx.x;
  int wave = tid >> 5, lane = tid & 31;
  int task = blockIdx.x * 4 + wave;
  int head = task / (NREF / 16);   // uniform across the block (256 % 4 == 0)
  int qt   = task % (NREF / 16);
  int r0 = qt * 16;
  int half = lane >> 4, cil = lane & 15;
  const v8f vzero = {};

  const __bf16* kg0 = qkv + EDIM + head * DH;          // K element (key,d) = kg0[key*QKVD + d]
  const __bf16* vg0 = vT + (size_t)head * DH * NREF;   // V^T element (d,key) = vg0[d*NREF + key]
  __bf16* P = ldsP[wave];
  unsigned kbase = (unsigned)(size_t)&ldsK[0];         // generic->LDS offset (low 32 bits)
  unsigned vbase = (unsigned)(size_t)&ldsV[0];

  v16bf qa0 = load_a_frag(qkv, QKVD, r0, head * DH);
  v16bf qa1 = load_a_frag(qkv, QKVD, r0, head * DH + 32);

  float m_r[8], l_r[8];
#pragma unroll
  for (int r = 0; r < 8; ++r) { m_r[r] = -1e30f; l_r[r] = 0.0f; }
  v8f o[4];
#pragma unroll
  for (int t = 0; t < 4; ++t) o[t] = vzero;

  for (int j = 0; j < NREF; j += 32) {
    // ---- stage K block [32x64] and V^T block [64x32] into LDS (async) ----
#pragma unroll
    for (int p = 0; p < 2; ++p) {
      int s = tid + p * 128;                 // 256 16B segments each
      int kkey = s >> 3, kd = (s & 7) * 8;   // K: 32 rows x 8 segs
      async_load_b128(kbase + (unsigned)((kkey * DH + kd) * 2),
                      kg0 + (size_t)(j + kkey) * QKVD + kd);
      int vd = s >> 2, vk = (s & 3) * 8;     // V^T: 64 rows x 4 segs
      async_load_b128(vbase + (unsigned)((vd * 32 + vk) * 2),
                      vg0 + (size_t)vd * NREF + j + vk);
    }
    wait_async0();
    __syncthreads();

    // ---- scores: 16x32 block of q . k^T / sqrt(DH) (K from LDS) ----
    v8f s[2];
#pragma unroll
    for (int t = 0; t < 2; ++t) {
      v16bf b0 = load_bT_frag(ldsK, DH, t * 16, 0);
      v16bf b1 = load_bT_frag(ldsK, DH, t * 16, 32);
      v8f c = vzero;
      c = wmma_bf16(qa0, b0, c);
      c = wmma_bf16(qa1, b1, c);
#pragma unroll
      for (int r = 0; r < 8; ++r) c[r] *= 0.125f;
      s[t] = c;
    }
    // ---- online softmax (reduce across 16-lane half) ----
    float mnew[8], alpha[8], rs[8];
#pragma unroll
    for (int r = 0; r < 8; ++r) {
      float v = fmaxf(s[0][r], s[1][r]);
      v = fmaxf(v, __shfl_xor(v, 1, 32));
      v = fmaxf(v, __shfl_xor(v, 2, 32));
      v = fmaxf(v, __shfl_xor(v, 4, 32));
      v = fmaxf(v, __shfl_xor(v, 8, 32));
      mnew[r] = fmaxf(m_r[r], v);
      alpha[r] = __expf(m_r[r] - mnew[r]);
      rs[r] = 0.0f;
    }
#pragma unroll
    for (int t = 0; t < 2; ++t) {
#pragma unroll
      for (int r = 0; r < 8; ++r) {
        float p = __expf(s[t][r] - mnew[r]);
        rs[r] += p;
        P[(r + 8 * half) * 32 + t * 16 + cil] = (__bf16)p;
      }
    }
#pragma unroll
    for (int r = 0; r < 8; ++r) {
      float v = rs[r];
      v += __shfl_xor(v, 1, 32);
      v += __shfl_xor(v, 2, 32);
      v += __shfl_xor(v, 4, 32);
      v += __shfl_xor(v, 8, 32);
      l_r[r] = l_r[r] * alpha[r] + v;
      m_r[r] = mnew[r];
    }
    __syncthreads();
    // ---- P [16x32] @ V [32x64] (V from LDS, contiguous in key) ----
    v16bf pa = load_a_frag(P, 32, 0, 0);
#pragma unroll
    for (int dt = 0; dt < 4; ++dt) {
#pragma unroll
      for (int r = 0; r < 8; ++r) o[dt][r] *= alpha[r];
      v16bf b = load_bT_frag(ldsV, 32, dt * 16, 0);
      o[dt] = wmma_bf16(pa, b, o[dt]);
    }
    __syncthreads();   // WAR: protect ldsK/ldsV/ldsP before next staging
  }

  // ctx = o / l, laid out [N, E] row-major bf16
#pragma unroll
  for (int dt = 0; dt < 4; ++dt) {
#pragma unroll
    for (int r = 0; r < 8; ++r) {
      int row = r0 + r + 8 * half;
      int col = head * DH + dt * 16 + cil;
      ctx[(size_t)row * EDIM + col] = (__bf16)(o[dt][r] / l_r[r]);
    }
  }
}

// ---------------------------------------------------------------------------
// Kernel 4: output projection + residual: x = ctx @ wo + bo + fused
// One wave -> 16x64 tile. blockDim=128, grid = 256*6/4
// ---------------------------------------------------------------------------
__global__ void proj_kernel(const __bf16* __restrict__ ctx,
                            const __bf16* __restrict__ woT,
                            const float* __restrict__ bo,
                            const float* __restrict__ fused_f32,
                            float* __restrict__ x) {
  int wave = threadIdx.x >> 5, lane = threadIdx.x & 31;
  int task = blockIdx.x * 4 + wave;
  int rt = task / 6, cc = task % 6;
  int r0 = rt * 16, c0 = cc * 64;
  int half = lane >> 4, cil = lane & 15;
  const v8f vzero = {};
  v8f acc[4];
#pragma unroll
  for (int t = 0; t < 4; ++t) acc[t] = vzero;
  for (int kc = 0; kc < EDIM; kc += 32) {
    v16bf a = load_a_frag(ctx, EDIM, r0, kc);
#pragma unroll
    for (int t = 0; t < 4; ++t) {
      v16bf b = load_bT_frag(woT, EDIM, c0 + t * 16, kc);
      acc[t] = wmma_bf16(a, b, acc[t]);
    }
  }
#pragma unroll
  for (int t = 0; t < 4; ++t) {
    int col = c0 + t * 16 + cil;
    float bs = bo[col];
#pragma unroll
    for (int r = 0; r < 8; ++r) {
      int row = r0 + r + 8 * half;
      size_t idx = (size_t)row * EDIM + col;
      x[idx] = acc[t][r] + bs + fused_f32[idx];
    }
  }
}

// ---------------------------------------------------------------------------
// Kernel 5: sum(ref_w) + zero res accumulator. One block of 512.
// ---------------------------------------------------------------------------
__global__ void prep_kernel(const float* __restrict__ ref_w,
                            float* __restrict__ sum_out,
                            float* __restrict__ res) {
  __shared__ float sm[512];
  int tid = threadIdx.x;
  float s = 0.0f;
  for (int i = tid; i < NREF; i += 512) s += ref_w[i];
  sm[tid] = s;
  __syncthreads();
  for (int off = 256; off > 0; off >>= 1) {
    if (tid < off) sm[tid] += sm[tid + off];
    __syncthreads();
  }
  if (tid == 0) sum_out[0] = sm[0];
  if (tid < EDIM) res[tid] = 0.0f;
}

// ---------------------------------------------------------------------------
// Kernel 6: LayerNorm per row + weighted accumulate into res[E].
// One wave per row. blockDim=256, grid = NREF/8
// ---------------------------------------------------------------------------
__global__ void ln_reduce_kernel(const float* __restrict__ x,
                                 const float* __restrict__ ref_w,
                                 const float* __restrict__ sum_ptr,
                                 const float* __restrict__ ln_g,
                                 const float* __restrict__ ln_b,
                                 float* __restrict__ res) {
  int wave = threadIdx.x >> 5, lane = threadIdx.x & 31;
  int row = blockIdx.x * 8 + wave;
  const float* xr = x + (size_t)row * EDIM;
  float vals[12];
  float s = 0.0f;
#pragma unroll
  for (int k = 0; k < 12; ++k) { vals[k] = xr[lane + 32 * k]; s += vals[k]; }
#pragma unroll
  for (int m = 1; m < 32; m <<= 1) s += __shfl_xor(s, m, 32);
  float mu = s * (1.0f / EDIM);
  float vs = 0.0f;
#pragma unroll
  for (int k = 0; k < 12; ++k) { float d = vals[k] - mu; vs += d * d; }
#pragma unroll
  for (int m = 1; m < 32; m <<= 1) vs += __shfl_xor(vs, m, 32);
  float inv = 1.0f / sqrtf(vs * (1.0f / EDIM) + 1e-5f);
  float w = ref_w[row] / sum_ptr[0];
#pragma unroll
  for (int k = 0; k < 12; ++k) {
    int col = lane + 32 * k;
    float xn = (vals[k] - mu) * inv * ln_g[col] + ln_b[col];
    atomicAdd(&res[col], xn * w);
  }
}

// ---------------------------------------------------------------------------
// Kernel 7: final L2 normalize. One block of 512.
// ---------------------------------------------------------------------------
__global__ void norm_kernel(const float* __restrict__ res,
                            float* __restrict__ out) {
  __shared__ float sm[512];
  int tid = threadIdx.x;
  float v = (tid < EDIM) ? res[tid] : 0.0f;
  sm[tid] = v * v;
  __syncthreads();
  for (int off = 256; off > 0; off >>= 1) {
    if (tid < off) sm[tid] += sm[tid + off];
    __syncthreads();
  }
  float nrm = sqrtf(sm[0]);
  if (tid < EDIM) out[tid] = v / (nrm + 1e-8f);
}

// ---------------------------------------------------------------------------
extern "C" void kernel_launch(void* const* d_in, const int* in_sizes, int n_in,
                              void* d_out, int out_size, void* d_ws, size_t ws_size,
                              hipStream_t stream) {
  const float* low       = (const float*)d_in[0];
  const float* high      = (const float*)d_in[1];
  const float* ref_w     = (const float*)d_in[2];
  const float* low_w1    = (const float*)d_in[3];
  const float* low_b1    = (const float*)d_in[4];
  const float* low_w2    = (const float*)d_in[5];
  const float* low_b2    = (const float*)d_in[6];
  const float* high_w1   = (const float*)d_in[7];
  const float* high_b1   = (const float*)d_in[8];
  const float* high_w2   = (const float*)d_in[9];
  const float* high_b2   = (const float*)d_in[10];
  const float* layer_fuse= (const float*)d_in[11];
  const float* wqkv      = (const float*)d_in[12];
  const float* bqkv      = (const float*)d_in[13];
  const float* wo        = (const float*)d_in[14];
  const float* bo        = (const float*)d_in[15];
  const float* ln_g      = (const float*)d_in[16];
  const float* ln_b      = (const float*)d_in[17];

  char* ws = (char*)d_ws;
  size_t off = 0;
  auto alloc = [&](size_t bytes) -> void* {
    void* p = ws + off;
    off += (bytes + 255) & ~(size_t)255;
    return p;
  };
  const size_t NE = (size_t)NREF * EDIM;
  __bf16* lowb     = (__bf16*)alloc(NE * 2);
  __bf16* highb    = (__bf16*)alloc(NE * 2);
  __bf16* w1lt     = (__bf16*)alloc((size_t)EDIM * MIDD * 2);
  __bf16* w2lt     = (__bf16*)alloc((size_t)MIDD * EDIM * 2);
  __bf16* w1ht     = (__bf16*)alloc((size_t)EDIM * MIDD * 2);
  __bf16* w2ht     = (__bf16*)alloc((size_t)MIDD * EDIM * 2);
  __bf16* wqkvt    = (__bf16*)alloc((size_t)EDIM * QKVD * 2);
  __bf16* wot      = (__bf16*)alloc((size_t)EDIM * EDIM * 2);
  float*  fused_f32= (float*)alloc(NE * 4);
  __bf16* fused_b16= (__bf16*)alloc(NE * 2);
  __bf16* qkv_b16  = (__bf16*)alloc((size_t)NREF * QKVD * 2);
  __bf16* vT_b16   = (__bf16*)alloc((size_t)HEADS * DH * NREF * 2);
  __bf16* ctx_b16  = (__bf16*)alloc(NE * 2);
  float*  x_f32    = (float*)alloc(NE * 4);
  float*  res      = (float*)alloc(EDIM * 4);
  float*  sumw     = (float*)alloc(256);

  auto cvt = [&](const float* src, __bf16* dst, int n) {
    cvt_bf16_kernel<<<(n + 255) / 256, 256, 0, stream>>>(src, dst, n);
  };
  auto cvtT = [&](const float* src, __bf16* dst, int K, int N) {
    int n = K * N;
    cvt_bf16_t_kernel<<<(n + 255) / 256, 256, 0, stream>>>(src, dst, K, N);
  };
  cvt(low,  lowb,  (int)NE);
  cvt(high, highb, (int)NE);
  cvtT(low_w1,  w1lt, EDIM, MIDD);
  cvtT(low_w2,  w2lt, MIDD, EDIM);
  cvtT(high_w1, w1ht, EDIM, MIDD);
  cvtT(high_w2, w2ht, MIDD, EDIM);
  cvtT(wqkv, wqkvt, EDIM, QKVD);
  cvtT(wo,   wot,   EDIM, EDIM);

  mlp_fuse_kernel<<<NREF / 32, 64, 0, stream>>>(
      low, high, lowb, highb,
      w1lt, low_b1, w2lt, low_b2,
      w1ht, high_b1, w2ht, high_b2,
      layer_fuse, fused_f32, fused_b16);

  qkv_kernel<<<(NREF / 16) * 18 / 4, 128, 0, stream>>>(fused_b16, wqkvt, bqkv,
                                                       qkv_b16, vT_b16);

  attn_kernel<<<HEADS * (NREF / 16) / 4, 128, 0, stream>>>(qkv_b16, vT_b16, ctx_b16);

  proj_kernel<<<(NREF / 16) * 6 / 4, 128, 0, stream>>>(ctx_b16, wot, bo, fused_f32, x_f32);

  prep_kernel<<<1, 512, 0, stream>>>(ref_w, sumw, res);

  ln_reduce_kernel<<<NREF / 8, 256, 0, stream>>>(x_f32, ref_w, sumw, ln_g, ln_b, res);

  norm_kernel<<<1, 512, 0, stream>>>(res, (float*)d_out);
}
